// Attention_25589415149836
// MI455X (gfx1250) — matrix operand order repaired and checked by reading, not compiled, not run on hardware
//
#include <hip/hip_runtime.h>
#include <hip/hip_bf16.h>
#include <math.h>

// MI455X / gfx1250, wave32.
// B=128, H=256, N=2048.  Memory-bound streaming GEMM + tanh + v-dot + softmax.

typedef __attribute__((ext_vector_type(16))) __bf16 v16bf;
typedef __attribute__((ext_vector_type(8)))  __bf16 v8bf;
typedef __attribute__((ext_vector_type(2)))  __bf16 v2bf;
typedef __attribute__((ext_vector_type(8)))  float  v8f;

#define NB   128      // batches
#define HH   256      // H
#define NN   2048     // N
#define K2   512      // 2H (static+dynamic contraction length)
#define W3H  768      // 3H (row stride of W)
#define NT   64       // N-tile per workgroup
#define KB   32       // K chunk (matches WMMA K for bf16)
#define LROW 40       // padded LDS row length in bf16 elems (80B: 16B-aligned rows)

// gfx1250 native transcendental tanh (V_TANH_F32)
static __device__ __forceinline__ float fast_tanh(float x) {
#if __has_builtin(__builtin_amdgcn_tanhf)
  return __builtin_amdgcn_tanhf(x);
#else
  float r;
  // trailing v_nop satisfies the 1-op TRANS result hazard (compiler can't see it)
  asm("v_tanh_f32 %0, %1\n\tv_nop" : "=v"(r) : "v"(x));
  return r;
#endif
}

// ---------------------------------------------------------------- prep: W -> bf16 [256][512]
__global__ __launch_bounds__(256) void prep_w(const float* __restrict__ W,
                                              __bf16* __restrict__ Wbf) {
  int idx = blockIdx.x * 256 + threadIdx.x;      // 0 .. 131071
  int h = idx >> 9;                              // /512
  int k = idx & 511;
  Wbf[idx] = (__bf16)W[h * W3H + k];
}

// ---------------------------------------------------------------- prep: bias[b,h] = W[h,512:768] . dec[b,:]
__global__ __launch_bounds__(256) void prep_bias(const float* __restrict__ W,
                                                 const float* __restrict__ dec,
                                                 float* __restrict__ bias) {
  int b = blockIdx.x;
  int h = threadIdx.x;
  const float* wr = W + h * W3H + K2;
  const float* dr = dec + b * HH;
  float a = 0.f;
#pragma unroll 8
  for (int j = 0; j < HH; ++j) a = fmaf(wr[j], dr[j], a);
  bias[b * HH + h] = a;
}

// ---------------------------------------------------------------- main: logits[b, n0:n0+64]
__global__ __launch_bounds__(256) void attn_main(const float* __restrict__ sh,
                                                 const float* __restrict__ dh,
                                                 const float* __restrict__ vvec,
                                                 const __bf16* __restrict__ Wbf,
                                                 const float* __restrict__ bias,
                                                 float* __restrict__ logits) {
  __shared__ __bf16 Alds[HH * LROW];   // W chunk,      [h][k]   20.0 KB
  __shared__ __bf16 Blds[NT * LROW];   // hidden chunk, [n][k]    5.0 KB
  __shared__ float  vlds[HH];
  __shared__ float  blds[HH];
  __shared__ float  loglds[NT];

  const int tid  = threadIdx.x;
  const int wave = tid >> 5;           // 0..7
  const int lane = tid & 31;
  const int half = lane >> 4;          // 0/1 lane group (ISA 7.12.2 layouts)
  const int ln   = lane & 15;
  const int b    = blockIdx.y;
  const int n0   = blockIdx.x * NT;

  vlds[tid] = vvec[tid];
  blds[tid] = bias[b * HH + tid];
  if (tid < NT) loglds[tid] = 0.f;

  // B staging map: thread -> k-pair (2*kp, 2*kp+1) x n-quad nq (coalesced global reads,
  // packed v2bf (= one v_cvt_pk_bf16_f32 + one ds_store_b32) per (n, k-pair))
  const int kp = tid >> 4;             // 0..15
  const int nq = tid & 15;             // float4 column index

  v8f acc[8];                          // [t(2)][c(4)] 16x16 f32 tiles
#pragma unroll
  for (int i = 0; i < 8; ++i)
    acc[i] = (v8f){0.f, 0.f, 0.f, 0.f, 0.f, 0.f, 0.f, 0.f};

  for (int k0 = 0; k0 < K2; k0 += KB) {
    __syncthreads();   // previous chunk consumed

    // --- stage A: W bf16 chunk [256 h x 32 k], 64B per thread (contiguous)
    {
      const uint4* ga = (const uint4*)(Wbf + (size_t)tid * K2 + k0);
      uint4* la = (uint4*)(&Alds[tid * LROW]);
      la[0] = ga[0]; la[1] = ga[1]; la[2] = ga[2]; la[3] = ga[3];
    }
    // --- stage B: hidden f32 chunk [32 k x 64 n] -> bf16 pairs, transposed to [n][k]
    {
      const float* src = (k0 < 256) ? sh : dh;
      const int kb = (k0 < 256) ? k0 : (k0 - 256);
      const float* base = src + ((size_t)(b * HH + kb + 2 * kp)) * NN + n0;
      const float4 f0 = ((const float4*)base)[nq];        // row k = 2*kp
      const float4 f1 = ((const float4*)(base + NN))[nq]; // row k = 2*kp+1
      const int n = nq * 4;
      v2bf* dst = (v2bf*)&Blds[0];     // v2bf stride per n-row: LROW/2 = 20
      dst[(n + 0) * 20 + kp] = (v2bf){(__bf16)f0.x, (__bf16)f1.x};
      dst[(n + 1) * 20 + kp] = (v2bf){(__bf16)f0.y, (__bf16)f1.y};
      dst[(n + 2) * 20 + kp] = (v2bf){(__bf16)f0.z, (__bf16)f1.z};
      dst[(n + 3) * 20 + kp] = (v2bf){(__bf16)f0.w, (__bf16)f1.w};
    }
    __syncthreads();

    // --- build A fragments (ISA 7.12.2: 16-bit A 16x32; lane<16: K 0-7,16-23; lane>=16: K 8-15,24-31)
    v16bf afr[2];
#pragma unroll
    for (int t = 0; t < 2; ++t) {
      int row = wave * 32 + t * 16 + ln;
      v8bf lo = *(const v8bf*)(&Alds[row * LROW + half * 8]);
      v8bf hi = *(const v8bf*)(&Alds[row * LROW + 16 + half * 8]);
      afr[t] = __builtin_shufflevector(lo, hi, 0, 1, 2, 3, 4, 5, 6, 7,
                                       8, 9, 10, 11, 12, 13, 14, 15);
    }
    // --- B fragments (dense B 32x16: lanes 0-15 K 0-7,16-23; lanes 16-31 K 8-15,24-31) + 8 WMMAs
#pragma unroll
    for (int c = 0; c < 4; ++c) {
      int col = c * 16 + ln;
      v8bf lo = *(const v8bf*)(&Blds[col * LROW + half * 8]);
      v8bf hi = *(const v8bf*)(&Blds[col * LROW + 16 + half * 8]);
      v16bf bfr = __builtin_shufflevector(lo, hi, 0, 1, 2, 3, 4, 5, 6, 7,
                                          8, 9, 10, 11, 12, 13, 14, 15);
      acc[0 * 4 + c] = __builtin_amdgcn_wmma_f32_16x16x32_bf16(
          false, afr[0], false, bfr, (short)0, acc[0 * 4 + c], false, false);
      acc[1 * 4 + c] = __builtin_amdgcn_wmma_f32_16x16x32_bf16(
          false, afr[1], false, bfr, (short)0, acc[1 * 4 + c], false, false);
    }
  }

  // --- epilogue: logits[n] += v[h] * tanh(S[h][n] + bias[h]), reduce over h via LDS atomics
#pragma unroll
  for (int c = 0; c < 4; ++c) {
    float part = 0.f;
#pragma unroll
    for (int t = 0; t < 2; ++t) {
#pragma unroll
      for (int r = 0; r < 8; ++r) {
        int M = wave * 32 + t * 16 + half * 8 + r;   // C-tile row map (ISA 7.12.2)
        part += vlds[M] * fast_tanh(acc[t * 4 + c][r] + blds[M]);
      }
    }
    atomicAdd(&loglds[c * 16 + ln], part);           // ds_add_f32
  }
  __syncthreads();
  if (tid < NT) logits[(size_t)b * NN + n0 + tid] = loglds[tid];
}

// ---------------------------------------------------------------- softmax (in-place on d_out rows)
__global__ __launch_bounds__(256) void softmax_rows(float* __restrict__ out) {
  __shared__ float red[8];
  const int b = blockIdx.x;
  const int tid = threadIdx.x;
  const int wave = tid >> 5;
  float* row = out + (size_t)b * NN;

  float v[8];
  float m = -3.4e38f;
#pragma unroll
  for (int i = 0; i < 8; ++i) { v[i] = row[tid + i * 256]; m = fmaxf(m, v[i]); }
#pragma unroll
  for (int off = 16; off; off >>= 1) m = fmaxf(m, __shfl_xor(m, off, 32));
  if ((tid & 31) == 0) red[wave] = m;
  __syncthreads();
  m = red[0];
#pragma unroll
  for (int w = 1; w < 8; ++w) m = fmaxf(m, red[w]);
  __syncthreads();

  float s = 0.f;
#pragma unroll
  for (int i = 0; i < 8; ++i) { v[i] = __expf(v[i] - m); s += v[i]; }
#pragma unroll
  for (int off = 16; off; off >>= 1) s += __shfl_xor(s, off, 32);
  if ((tid & 31) == 0) red[wave] = s;
  __syncthreads();
  s = 0.f;
#pragma unroll
  for (int w = 0; w < 8; ++w) s += red[w];
  const float inv = 1.f / s;
#pragma unroll
  for (int i = 0; i < 8; ++i) row[tid + i * 256] = v[i] * inv;
}

// ----------------------------------------------------------------
extern "C" void kernel_launch(void* const* d_in, const int* in_sizes, int n_in,
                              void* d_out, int out_size, void* d_ws, size_t ws_size,
                              hipStream_t stream) {
  const float* sh  = (const float*)d_in[0];   // static_hidden  [128,256,2048]
  const float* dh  = (const float*)d_in[1];   // dynamic_hidden [128,256,2048]
  const float* dec = (const float*)d_in[2];   // decoder_hidden [128,256]
  const float* v   = (const float*)d_in[3];   // v [1,1,256]
  const float* W   = (const float*)d_in[4];   // W [1,256,768]
  float* out = (float*)d_out;                 // [128,1,2048]

  __bf16* Wbf  = (__bf16*)d_ws;                               // 256*512*2 = 256 KB
  float*  bias = (float*)((char*)d_ws + (size_t)HH * K2 * 2); // 128*256*4 = 128 KB
  (void)in_sizes; (void)n_in; (void)out_size; (void)ws_size;

  prep_w   <<<512, 256, 0, stream>>>(W, Wbf);
  prep_bias<<<NB,  256, 0, stream>>>(W, dec, bias);
  attn_main<<<dim3(NN / NT, NB), 256, 0, stream>>>(sh, dh, v, Wbf, bias, out);
  softmax_rows<<<NB, 256, 0, stream>>>(out);
}
